// Net_16466904613603
// MI455X (gfx1250) — matrix operand order repaired and checked by reading, not compiled, not run on hardware
//
#include <hip/hip_runtime.h>
#include <hip/hip_bf16.h>
#include <math.h>

#define B_  64
#define N_  207
#define T_  12
#define F_  2
#define H_  64
#define E_  1722
#define BN_ (B_*N_)    // 13248
#define EB_ (E_*B_)    // 110208

typedef __bf16 v16bf __attribute__((ext_vector_type(16)));
typedef float  v8f   __attribute__((ext_vector_type(8)));

// ---------------------------------------------------------------------------
// Weight packing: row-major fp32 W[Nout,Kact] -> bf16 packed in the WMMA
// B-matrix (KxN) lane layout.  packed[(((nt*ksteps)+ks)*32 + lane)*16 + e]
//   = W[nt*16 + lane%16][ks*32 + (lane/16)*16 + e]   (0 beyond Kact)
// ---------------------------------------------------------------------------
__global__ __launch_bounds__(256) void pack_w_kernel(
    const float* __restrict__ W, unsigned short* __restrict__ dst,
    int Kact, int Kpad, int Nout) {
  int idx = blockIdx.x * blockDim.x + threadIdx.x;
  int ksteps = Kpad >> 5;
  int total = (Nout >> 4) * ksteps * 512;
  if (idx >= total) return;
  int e    = idx & 15;
  int l    = (idx >> 4) & 31;
  int rest = idx >> 9;
  int ks   = rest % ksteps;
  int nt   = rest / ksteps;
  int n = nt * 16 + (l & 15);
  int k = ks * 32 + ((l >> 4) << 4) + e;
  float v = (k < Kact) ? W[(size_t)n * Kact + k] : 0.0f;
  __bf16 bv = (__bf16)v;
  dst[idx] = __builtin_bit_cast(unsigned short, bv);
}

// Load 16 contiguous fp32 (k = 16*half + s, same order as packed B) and convert
// to the v16bf A fragment for one 16x32 tile row.
__device__ __forceinline__ v16bf load_a_frag(const float* __restrict__ p0) {
  const float* p1 = p0 + 8;
  float4 f0 = ((const float4*)p0)[0];
  float4 f1 = ((const float4*)p0)[1];
  float4 f2 = ((const float4*)p1)[0];
  float4 f3 = ((const float4*)p1)[1];
  v16bf a;
  a[0]=(__bf16)f0.x; a[1]=(__bf16)f0.y; a[2]=(__bf16)f0.z; a[3]=(__bf16)f0.w;
  a[4]=(__bf16)f1.x; a[5]=(__bf16)f1.y; a[6]=(__bf16)f1.z; a[7]=(__bf16)f1.w;
  a[8]=(__bf16)f2.x; a[9]=(__bf16)f2.y; a[10]=(__bf16)f2.z; a[11]=(__bf16)f2.w;
  a[12]=(__bf16)f3.x; a[13]=(__bf16)f3.y; a[14]=(__bf16)f3.z; a[15]=(__bf16)f3.w;
  return a;
}

// ---------------------------------------------------------------------------
// GEMM: Y[M,Nout] = act( X[M,Kpad](fp32,row-major,zero-padded) * Wt + bias )
// One wave computes a 32x64 output block (2 M-tiles x 4 N-tiles), K-loop in
// steps of 32 via v_wmma_f32_16x16x32_bf16 (fp32 accumulation).
// A fragments reused across 4 N-tiles; B fragments reused across 2 M-tiles.
// ---------------------------------------------------------------------------
__global__ __launch_bounds__(256) void gemm_bf16_wmma(
    const float* __restrict__ X, const unsigned short* __restrict__ Bp,
    const float* __restrict__ bias, float* __restrict__ Y,
    int M, int Kpad, int Nout, int relu) {
  int wave = blockIdx.x * (blockDim.x >> 5) + (threadIdx.x >> 5);
  int nblk = Nout >> 6;                 // 64-wide N blocks
  int mb = wave / nblk;                 // 32-high M block
  int nb = wave % nblk;
  if (mb >= (M >> 5)) return;

  int lane = threadIdx.x & 31;
  int half = lane >> 4;
  int l16  = lane & 15;
  int ksteps = Kpad >> 5;

  const float* xrow0 = X + (size_t)(mb * 32 + l16) * Kpad;
  const float* xrow1 = xrow0 + (size_t)16 * Kpad;
  const unsigned short* bp =
      Bp + ((size_t)(nb * 4) * ksteps * 32 + lane) * 16;
  size_t btile = (size_t)ksteps * 512;  // packed elems per 16-col tile

  v8f acc[2][4];
#pragma unroll
  for (int i = 0; i < 2; ++i)
#pragma unroll
    for (int j = 0; j < 4; ++j)
      acc[i][j] = (v8f){0.f, 0.f, 0.f, 0.f, 0.f, 0.f, 0.f, 0.f};

  for (int ks = 0; ks < ksteps; ++ks) {
    int koff = ks * 32 + half * 16;
    v16bf a0 = load_a_frag(xrow0 + koff);
    v16bf a1 = load_a_frag(xrow1 + koff);
    const unsigned short* bks = bp + (size_t)ks * 512;
#pragma unroll
    for (int j = 0; j < 4; ++j) {
      v16bf b = *(const v16bf*)(bks + (size_t)j * btile);
      acc[0][j] = __builtin_amdgcn_wmma_f32_16x16x32_bf16(
          false, a0, false, b, (short)0, acc[0][j], false, false);
      acc[1][j] = __builtin_amdgcn_wmma_f32_16x16x32_bf16(
          false, a1, false, b, (short)0, acc[1][j], false, false);
    }
  }

#pragma unroll
  for (int j = 0; j < 4; ++j) {
    int col = (nb * 4 + j) * 16 + l16;
    float bv = bias[col];
#pragma unroll
    for (int i = 0; i < 2; ++i) {
      int rbase = mb * 32 + i * 16 + (half ? 8 : 0);
#pragma unroll
      for (int r = 0; r < 8; ++r) {
        float v = acc[i][j][r] + bv;
        if (relu) v = fmaxf(v, 0.f);
        Y[(size_t)(rbase + r) * Nout + col] = v;
      }
    }
  }
}

// ---------------------------------------------------------------------------
// Elementwise / gather / scatter kernels
// ---------------------------------------------------------------------------
__global__ __launch_bounds__(256) void fill_zero_kernel(float* __restrict__ p, long long n) {
  long long i = (long long)blockIdx.x * blockDim.x + threadIdx.x;
  if (i < n) p[i] = 0.f;
}

// Encoder GRU gate: folds xW = x*Wih^T + bih inline (F=2).  h updated in place.
__global__ __launch_bounds__(256) void enc_gate_kernel(
    const float* __restrict__ x, const float* __restrict__ Wih,
    const float* __restrict__ bih, const float* __restrict__ hW,
    float* __restrict__ h, int t) {
  int idx = blockIdx.x * blockDim.x + threadIdx.x;
  if (idx >= BN_ * H_) return;
  int row = idx / H_, hh = idx % H_;
  int b = row / N_, n = row % N_;
  const float* xp = x + (((size_t)b * T_ + t) * N_ + n) * F_;
  float x0 = xp[0], x1v = xp[1];
  float xw[3], hw[3];
#pragma unroll
  for (int g = 0; g < 3; ++g) {
    int j = g * H_ + hh;
    xw[g] = bih[j] + x0 * Wih[j * F_ + 0] + x1v * Wih[j * F_ + 1];
    hw[g] = hW[(size_t)row * (3 * H_) + j];
  }
  float r  = 1.f / (1.f + expf(-(xw[0] + hw[0])));
  float z  = 1.f / (1.f + expf(-(xw[1] + hw[1])));
  float nn = tanhf(xw[2] + r * hw[2]);
  h[idx] = (1.f - z) * nn + z * h[idx];
}

// Decoder GRU gate (H2=128); y_in[t] = t==0 ? x[:,T-1] : y[:,t-1]; writes hs.
__global__ __launch_bounds__(256) void dec_gate_kernel(
    const float* __restrict__ x, const float* __restrict__ y,
    const float* __restrict__ Wih, const float* __restrict__ bih,
    const float* __restrict__ hW, float* __restrict__ h,
    float* __restrict__ hs, int t) {
  const int H2 = 2 * H_;
  int idx = blockIdx.x * blockDim.x + threadIdx.x;
  if (idx >= BN_ * H2) return;
  int row = idx / H2, hh = idx % H2;
  int b = row / N_, n = row % N_;
  const float* sp = (t == 0) ? (x + (((size_t)b * T_ + (T_ - 1)) * N_ + n) * F_)
                             : (y + (((size_t)b * T_ + (t - 1)) * N_ + n) * F_);
  float x0 = sp[0], x1v = sp[1];
  float xw[3], hw[3];
#pragma unroll
  for (int g = 0; g < 3; ++g) {
    int j = g * H2 + hh;
    xw[g] = bih[j] + x0 * Wih[j * F_ + 0] + x1v * Wih[j * F_ + 1];
    hw[g] = hW[(size_t)row * (3 * H2) + j];
  }
  float r  = 1.f / (1.f + expf(-(xw[0] + hw[0])));
  float z  = 1.f / (1.f + expf(-(xw[1] + hw[1])));
  float nn = tanhf(xw[2] + r * hw[2]);
  float hn = (1.f - z) * nn + z * h[idx];
  h[idx] = hn;
  hs[(size_t)t * BN_ * H2 + idx] = hn;
}

// GN1 edge concat: [xs(64) xd(64) ea(1) u=0(64) pad(31)] -> [EB,224]
__global__ __launch_bounds__(256) void edge_concat1_kernel(
    const float* __restrict__ henc, const int* __restrict__ srcI,
    const int* __restrict__ dstI, const float* __restrict__ eattr,
    float* __restrict__ out) {
  const int K = 224;
  long long idx = (long long)blockIdx.x * blockDim.x + threadIdx.x;
  if (idx >= (long long)EB_ * K) return;
  int c = (int)(idx % K);
  int rowi = (int)(idx / K);
  int e = rowi / B_, b = rowi % B_;
  float v;
  if      (c < 64)   v = henc[((size_t)b * N_ + srcI[e]) * H_ + c];
  else if (c < 128)  v = henc[((size_t)b * N_ + dstI[e]) * H_ + (c - 64)];
  else if (c == 128) v = eattr[e];
  else               v = 0.f;          // u == 0 at layer 1, plus zero pad
  out[idx] = v;
}

// GN2 edge concat: [xs(128) xd(128) ea1(128) u1(128)] -> [EB,512]
__global__ __launch_bounds__(256) void edge_concat2_kernel(
    const float* __restrict__ x1, const float* __restrict__ ea1,
    const float* __restrict__ u1, const int* __restrict__ srcI,
    const int* __restrict__ dstI, float* __restrict__ out) {
  const int K = 512;
  long long idx = (long long)blockIdx.x * blockDim.x + threadIdx.x;
  if (idx >= (long long)EB_ * K) return;
  int c = (int)(idx % K);
  int rowi = (int)(idx / K);
  int e = rowi / B_, b = rowi % B_;
  float v;
  if      (c < 128) v = x1[((size_t)srcI[e] * B_ + b) * 128 + c];
  else if (c < 256) v = x1[((size_t)dstI[e] * B_ + b) * 128 + (c - 128)];
  else if (c < 384) v = ea1[((size_t)e * B_ + b) * 128 + (c - 256)];
  else              v = u1[(size_t)b * 128 + (c - 384)];
  out[idx] = v;
}

__global__ __launch_bounds__(256) void scatter_add_kernel(
    const float* __restrict__ ea, const int* __restrict__ dstI,
    float* __restrict__ agg) {
  long long idx = (long long)blockIdx.x * blockDim.x + threadIdx.x;
  if (idx >= (long long)EB_ * 128) return;
  int e = (int)(idx / (B_ * 128));
  int r = (int)(idx % (B_ * 128));
  int b = r / 128, c = r % 128;
  atomicAdd(&agg[((size_t)dstI[e] * B_ + b) * 128 + c], ea[idx]);
}

// GN1 node concat: [x(64) agg(128) u=0(64)] -> [BN,256], rows n*B+b
__global__ __launch_bounds__(256) void node_concat1_kernel(
    const float* __restrict__ henc, const float* __restrict__ agg,
    float* __restrict__ out) {
  const int K = 256;
  long long idx = (long long)blockIdx.x * blockDim.x + threadIdx.x;
  if (idx >= (long long)BN_ * K) return;
  int c = (int)(idx % K);
  int rowi = (int)(idx / K);
  int n = rowi / B_, b = rowi % B_;
  float v;
  if      (c < 64)  v = henc[((size_t)b * N_ + n) * H_ + c];
  else if (c < 192) v = agg[(size_t)rowi * 128 + (c - 64)];
  else              v = 0.f;
  out[idx] = v;
}

// GN2 node concat: [x1(128) agg2(128) u1(128)] -> [BN,384]
__global__ __launch_bounds__(256) void node_concat2_kernel(
    const float* __restrict__ x1, const float* __restrict__ agg,
    const float* __restrict__ u1, float* __restrict__ out) {
  const int K = 384;
  long long idx = (long long)blockIdx.x * blockDim.x + threadIdx.x;
  if (idx >= (long long)BN_ * K) return;
  int c = (int)(idx % K);
  int rowi = (int)(idx / K);
  int b = rowi % B_;
  float v;
  if      (c < 128) v = x1[(size_t)rowi * 128 + c];
  else if (c < 256) v = agg[(size_t)rowi * 128 + (c - 128)];
  else              v = u1[(size_t)b * 128 + (c - 256)];
  out[idx] = v;
}

__global__ __launch_bounds__(256) void mean_nodes_kernel(
    const float* __restrict__ xin, float* __restrict__ xm) {
  int idx = blockIdx.x * blockDim.x + threadIdx.x;
  if (idx >= B_ * 128) return;
  int b = idx / 128, c = idx % 128;
  float s = 0.f;
  for (int n = 0; n < N_; ++n) s += xin[((size_t)n * B_ + b) * 128 + c];
  xm[idx] = s * (1.f / (float)N_);
}

// gn1_glob input: [u=0(64) mean(128)] -> [64,192]
__global__ __launch_bounds__(256) void glob_concat_kernel(
    const float* __restrict__ xm, float* __restrict__ out) {
  int idx = blockIdx.x * blockDim.x + threadIdx.x;
  if (idx >= B_ * 192) return;
  int b = idx / 192, c = idx % 192;
  out[idx] = (c < 64) ? 0.f : xm[(size_t)b * 128 + (c - 64)];
}

// h0_dec = concat(h_enc[b*N+n], graph_enc) rows b*N+n -> [BN,128]
__global__ __launch_bounds__(256) void h0_concat_kernel(
    const float* __restrict__ henc, const float* __restrict__ gout,
    float* __restrict__ hdec) {
  long long idx = (long long)blockIdx.x * blockDim.x + threadIdx.x;
  if (idx >= (long long)BN_ * 128) return;
  int c = (int)(idx % 128);
  int row = (int)(idx / 128);          // b*N + n
  int b = row / N_, n = row % N_;
  float v;
  if (c < 64) v = henc[(size_t)row * H_ + c];
  else        v = gout[((size_t)n * B_ + b) * H_ + (c - 64)];
  hdec[idx] = v;
}

__global__ __launch_bounds__(256) void out_proj_kernel(
    const float* __restrict__ hs, const float* __restrict__ Wout,
    const float* __restrict__ bout, float* __restrict__ out) {
  int idx = blockIdx.x * blockDim.x + threadIdx.x;
  if (idx >= T_ * BN_) return;
  int t = idx / BN_, row = idx % BN_;
  int b = row / N_, n = row % N_;
  const float* hp = hs + (size_t)idx * 128;
  float s = bout[0];
#pragma unroll 4
  for (int c = 0; c < 128; ++c) s += hp[c] * Wout[c];
  out[((size_t)b * T_ + t) * N_ + n] = s;
}

// ---------------------------------------------------------------------------
extern "C" void kernel_launch(void* const* d_in, const int* in_sizes, int n_in,
                              void* d_out, int out_size, void* d_ws, size_t ws_size,
                              hipStream_t stream) {
  const float* x      = (const float*)d_in[0];
  const float* y      = (const float*)d_in[1];
  const int*   ei     = (const int*)d_in[2];
  const float* eattr  = (const float*)d_in[3];
  const float* encWih = (const float*)d_in[4];
  const float* encWhh = (const float*)d_in[5];
  const float* encbih = (const float*)d_in[6];
  const float* encbhh = (const float*)d_in[7];
  const float* decWih = (const float*)d_in[8];
  const float* decWhh = (const float*)d_in[9];
  const float* decbih = (const float*)d_in[10];
  const float* decbhh = (const float*)d_in[11];
  const float* g1eW1 = (const float*)d_in[12]; const float* g1eb1 = (const float*)d_in[13];
  const float* g1eW2 = (const float*)d_in[14]; const float* g1eb2 = (const float*)d_in[15];
  const float* g1nW1 = (const float*)d_in[16]; const float* g1nb1 = (const float*)d_in[17];
  const float* g1nW2 = (const float*)d_in[18]; const float* g1nb2 = (const float*)d_in[19];
  const float* g1gW1 = (const float*)d_in[20]; const float* g1gb1 = (const float*)d_in[21];
  const float* g1gW2 = (const float*)d_in[22]; const float* g1gb2 = (const float*)d_in[23];
  const float* g2eW1 = (const float*)d_in[24]; const float* g2eb1 = (const float*)d_in[25];
  const float* g2eW2 = (const float*)d_in[26]; const float* g2eb2 = (const float*)d_in[27];
  const float* g2nW1 = (const float*)d_in[28]; const float* g2nb1 = (const float*)d_in[29];
  const float* g2nW2 = (const float*)d_in[30]; const float* g2nb2 = (const float*)d_in[31];
  // d_in[32..35] = gn2_glob (dead: does not affect the output) -- skipped
  const float* nodeW = (const float*)d_in[36]; const float* nodeb = (const float*)d_in[37];
  const float* outW  = (const float*)d_in[38]; const float* outb  = (const float*)d_in[39];

  const int* srcI = ei;        // edge_index[0]
  const int* dstI = ei + E_;   // edge_index[1]

  // ---- workspace arena ----
  size_t off = 0;
  auto alloc_f = [&](size_t n) -> float* {
    size_t o = (off + 255) & ~(size_t)255;
    off = o + n * sizeof(float);
    return (float*)((char*)d_ws + o);
  };
  auto alloc_bf = [&](size_t n) -> unsigned short* {
    size_t o = (off + 255) & ~(size_t)255;
    off = o + n * sizeof(unsigned short);
    return (unsigned short*)((char*)d_ws + o);
  };

  float* henc    = alloc_f((size_t)BN_ * H_);
  float* hw      = alloc_f((size_t)BN_ * 384);
  float* edgein  = alloc_f((size_t)EB_ * 512);
  float* hs      = edgein;                       // alias: decode phase only
  float* edgehid = alloc_f((size_t)EB_ * 256);
  float* hdec    = edgehid;                      // alias: decode phase only
  float* ea1     = alloc_f((size_t)EB_ * 128);   // also reused for new_ea2
  float* agg     = alloc_f((size_t)BN_ * 128);
  float* nodein  = alloc_f((size_t)BN_ * 384);
  float* nodehid = alloc_f((size_t)BN_ * 256);
  float* x1      = alloc_f((size_t)BN_ * 128);
  float* x2      = alloc_f((size_t)BN_ * 128);
  float* xmean   = alloc_f((size_t)B_ * 128);
  float* globin  = alloc_f((size_t)B_ * 192);
  float* globhid = alloc_f((size_t)B_ * 256);
  float* u1      = alloc_f((size_t)B_ * 128);
  float* gout    = alloc_f((size_t)BN_ * H_);

  unsigned short* p_encWhh = alloc_bf((size_t)192 * 64);
  unsigned short* p_decWhh = alloc_bf((size_t)384 * 128);
  unsigned short* p_g1eW1  = alloc_bf((size_t)256 * 224);
  unsigned short* p_g1eW2  = alloc_bf((size_t)128 * 256);
  unsigned short* p_g1nW1  = alloc_bf((size_t)256 * 256);
  unsigned short* p_g1nW2  = alloc_bf((size_t)128 * 256);
  unsigned short* p_g1gW1  = alloc_bf((size_t)256 * 192);
  unsigned short* p_g1gW2  = alloc_bf((size_t)128 * 256);
  unsigned short* p_g2eW1  = alloc_bf((size_t)256 * 512);
  unsigned short* p_g2eW2  = alloc_bf((size_t)128 * 256);
  unsigned short* p_g2nW1  = alloc_bf((size_t)256 * 384);
  unsigned short* p_g2nW2  = alloc_bf((size_t)128 * 256);
  unsigned short* p_nodeW  = alloc_bf((size_t)64 * 128);

  auto cdiv = [](long long a, long long b) { return (int)((a + b - 1) / b); };
  auto PACK = [&](const float* W, int Kact, int Kpad, int Nout, unsigned short* dst) {
    long long tot = (long long)(Nout / 16) * (Kpad / 32) * 512;
    pack_w_kernel<<<cdiv(tot, 256), 256, 0, stream>>>(W, dst, Kact, Kpad, Nout);
  };
  auto GEMM = [&](const float* X, const unsigned short* Bp, const float* bias,
                  float* Y, int M, int Kpad, int Nout, int relu) {
    long long waves = (long long)(M / 32) * (Nout / 64);
    gemm_bf16_wmma<<<cdiv(waves, 8), 256, 0, stream>>>(X, Bp, bias, Y, M, Kpad, Nout, relu);
  };
  auto FILL = [&](float* p, long long n) {
    fill_zero_kernel<<<cdiv(n, 256), 256, 0, stream>>>(p, n);
  };

  // ---- pack all GEMM weights to bf16 B-layout ----
  PACK(encWhh, 64, 64, 192, p_encWhh);
  PACK(decWhh, 128, 128, 384, p_decWhh);
  PACK(g1eW1, 193, 224, 256, p_g1eW1);  PACK(g1eW2, 256, 256, 128, p_g1eW2);
  PACK(g1nW1, 256, 256, 256, p_g1nW1);  PACK(g1nW2, 256, 256, 128, p_g1nW2);
  PACK(g1gW1, 192, 192, 256, p_g1gW1);  PACK(g1gW2, 256, 256, 128, p_g1gW2);
  PACK(g2eW1, 512, 512, 256, p_g2eW1);  PACK(g2eW2, 256, 256, 128, p_g2eW2);
  PACK(g2nW1, 384, 384, 256, p_g2nW1);  PACK(g2nW2, 256, 256, 128, p_g2nW2);
  PACK(nodeW, 128, 128, 64, p_nodeW);

  // ---- encoder GRU (h starts at 0) ----
  FILL(henc, (long long)BN_ * H_);
  for (int t = 0; t < T_; ++t) {
    GEMM(henc, p_encWhh, encbhh, hw, BN_, 64, 192, 0);
    enc_gate_kernel<<<cdiv((long long)BN_ * H_, 256), 256, 0, stream>>>(
        x, encWih, encbih, hw, henc, t);
  }

  // ---- GN layer 1 ----
  edge_concat1_kernel<<<cdiv((long long)EB_ * 224, 256), 256, 0, stream>>>(
      henc, srcI, dstI, eattr, edgein);
  GEMM(edgein, p_g1eW1, g1eb1, edgehid, EB_, 224, 256, 1);
  GEMM(edgehid, p_g1eW2, g1eb2, ea1, EB_, 256, 128, 0);
  FILL(agg, (long long)BN_ * 128);
  scatter_add_kernel<<<cdiv((long long)EB_ * 128, 256), 256, 0, stream>>>(ea1, dstI, agg);
  node_concat1_kernel<<<cdiv((long long)BN_ * 256, 256), 256, 0, stream>>>(henc, agg, nodein);
  GEMM(nodein, p_g1nW1, g1nb1, nodehid, BN_, 256, 256, 1);
  GEMM(nodehid, p_g1nW2, g1nb2, x1, BN_, 256, 128, 0);
  mean_nodes_kernel<<<cdiv(B_ * 128, 256), 256, 0, stream>>>(x1, xmean);
  glob_concat_kernel<<<cdiv(B_ * 192, 256), 256, 0, stream>>>(xmean, globin);
  GEMM(globin, p_g1gW1, g1gb1, globhid, B_, 192, 256, 1);
  GEMM(globhid, p_g1gW2, g1gb2, u1, B_, 256, 128, 0);

  // ---- GN layer 2 ----
  edge_concat2_kernel<<<cdiv((long long)EB_ * 512, 256), 256, 0, stream>>>(
      x1, ea1, u1, srcI, dstI, edgein);
  GEMM(edgein, p_g2eW1, g2eb1, edgehid, EB_, 512, 256, 1);
  GEMM(edgehid, p_g2eW2, g2eb2, ea1 /* new_ea2 */, EB_, 256, 128, 0);
  FILL(agg, (long long)BN_ * 128);
  scatter_add_kernel<<<cdiv((long long)EB_ * 128, 256), 256, 0, stream>>>(ea1, dstI, agg);
  node_concat2_kernel<<<cdiv((long long)BN_ * 384, 256), 256, 0, stream>>>(x1, agg, u1, nodein);
  GEMM(nodein, p_g2nW1, g2nb1, nodehid, BN_, 384, 256, 1);
  GEMM(nodehid, p_g2nW2, g2nb2, x2, BN_, 256, 128, 0);
  GEMM(x2, p_nodeW, nodeb, gout, BN_, 128, 64, 0);

  // ---- decoder GRU ----
  h0_concat_kernel<<<cdiv((long long)BN_ * 128, 256), 256, 0, stream>>>(henc, gout, hdec);
  for (int t = 0; t < T_; ++t) {
    GEMM(hdec, p_decWhh, decbhh, hw, BN_, 128, 384, 0);
    dec_gate_kernel<<<cdiv((long long)BN_ * 256, 256), 256, 0, stream>>>(
        x, y, decWih, decbih, hw, hdec, hs, t);
  }
  out_proj_kernel<<<cdiv(T_ * BN_, 256), 256, 0, stream>>>(hs, outW, outb, (float*)d_out);
}
